// XPainnUpdate_29154238005843
// MI455X (gfx1250) — compile-verified, matrix operand order checked by reading, not run on hardware
//
#include <hip/hip_runtime.h>
#include <hip/hip_bf16.h>
#include <math.h>

// ---------------------------------------------------------------------------
// XPainnUpdate fused kernel for MI455X (gfx1250, wave32, WMMA 16x16x32 f16).
// One block = 16 nodes. All matmuls via v_wmma_f32_16x16x32_f16 (f32 accum).
// Weights are pre-converted once per launch to transposed f16 in d_ws.
// All WMMA k-chains fully unrolled so loads clause and overlap the MMA chain.
// ---------------------------------------------------------------------------

typedef __attribute__((ext_vector_type(16))) _Float16 v16h;
typedef __attribute__((ext_vector_type(8)))  float    v8f;

#define NM0   128
#define NM1   64
#define NM2   32
#define NIR   224     // M0+M1+M2
#define NDIM  128     // NODE_DIM
#define SPH   480     // M0 + 3*M1 + 5*M2
#define HID   480     // 2*NODE_DIM + NUM_IRREPS
#define KMLP  352     // NODE_DIM + NUM_IRREPS

// LDS layout (bytes)
#define OFF_ASPH   0u       // f16 16x480 normalized spherical (planar comps)
#define OFF_MLPA   15360u   // f16 16x352 [scalar_in | inv]
#define OFF_DOTA   26624u   // f16 16x224 dot features
#define OFF_HA     33792u   // f16 16x128 hidden (post-silu)
#define OFF_U      37888u   // f32 16x480 U (planar comps)
#define OFF_V      68608u   // f32 16x480 V
#define OFF_INNER  99328u   // f32 16x128 inner = dot @ dot_w
#define OFF_MOUT   107520u  // f32 16x480 mlp_out
#define SMEM_BYTES 138240u

#define WMMA_F16(a, b, c) \
  __builtin_amdgcn_wmma_f32_16x16x32_f16(false, (a), false, (b), (short)0, (c), false, false)

__global__ void cvt_transpose_f16(const float* __restrict__ src,
                                  _Float16* __restrict__ dst,
                                  int rows, int cols, float scale) {
  int i = blockIdx.x * 256 + threadIdx.x;
  if (i < rows * cols) {
    int r = i / cols, c = i % cols;
    dst[(size_t)c * rows + r] = (_Float16)(src[i] * scale);
  }
}

__device__ __forceinline__ float grp16_sum(float v) {
  v += __shfl_xor(v, 8, 16);
  v += __shfl_xor(v, 4, 16);
  v += __shfl_xor(v, 2, 16);
  v += __shfl_xor(v, 1, 16);
  return v;
}

__global__ void __launch_bounds__(256)
xpainn_fused(const float* __restrict__ xs, const float* __restrict__ xv,
             const float* __restrict__ ln_g, const float* __restrict__ ln_b,
             const float* __restrict__ o3w,  const float* __restrict__ o3b,
             const float* __restrict__ Ub0,  const float* __restrict__ Vb0,
             const float* __restrict__ mb1,  const float* __restrict__ mb2,
             const _Float16* __restrict__ Uw0T, const _Float16* __restrict__ Uw1T,
             const _Float16* __restrict__ Uw2T,
             const _Float16* __restrict__ Vw0T, const _Float16* __restrict__ Vw1T,
             const _Float16* __restrict__ Vw2T,
             const _Float16* __restrict__ dotwT, const _Float16* __restrict__ mw1T,
             const _Float16* __restrict__ mw2T,
             float* __restrict__ out_s, float* __restrict__ out_v) {
  extern __shared__ char smem[];
  _Float16* Asph  = (_Float16*)(smem + OFF_ASPH);
  _Float16* mlpA  = (_Float16*)(smem + OFF_MLPA);
  _Float16* dotA  = (_Float16*)(smem + OFF_DOTA);
  _Float16* hA    = (_Float16*)(smem + OFF_HA);
  float*    Ubuf  = (float*)(smem + OFF_U);
  float*    Vbuf  = (float*)(smem + OFF_V);
  float*    innerB= (float*)(smem + OFF_INNER);
  float*    moutB = (float*)(smem + OFF_MOUT);

  const int tid  = threadIdx.x;
  const int lane = tid & 31;
  const int wave = tid >> 5;
  const int row  = tid >> 4;   // 0..15 (16 threads per node-row)
  const int tt   = tid & 15;
  const size_t rowg = (size_t)blockIdx.x * 16 + row;
  const int mrow = (lane >> 4) * 8;  // WMMA C-layout: M = j + 8*(lane/16)
  const int alo  = (lane & 15);      // A/B row|col within tile
  const int khalf = (lane >> 4) << 4;

  // Warm L2 for the shared f16 weight image (global_prefetch_b8 path).
  __builtin_prefetch((const char*)Uw0T + tid * 1392, 0, 3);

  // ---------------- Phase A: layernorms -> f16 A tiles --------------------
  {
    const float* xr = xs + rowg * NDIM;
    float v0[8];
    float s = 0.f, sq = 0.f;
#pragma unroll
    for (int j = 0; j < 8; ++j) { float x = xr[tt * 8 + j]; v0[j] = x; s += x; sq += x * x; }
    s = grp16_sum(s); sq = grp16_sum(sq);
    float mu  = s * (1.f / NDIM);
    float var = sq * (1.f / NDIM) - mu * mu;
    float rs  = rsqrtf(var + 1e-5f);
#pragma unroll
    for (int j = 0; j < 8; ++j) {
      int c = tt * 8 + j;
      mlpA[row * KMLP + c] = (_Float16)((v0[j] - mu) * rs * ln_g[c] + ln_b[c]);
    }

    const float* vr = xv + rowg * SPH;
    float a0[8], a1[12], a2[10];
    float s0s = 0.f, s0q = 0.f, s1q = 0.f, s2q = 0.f;
#pragma unroll
    for (int j = 0; j < 8; ++j)  { float x = vr[tt * 8  + j];       a0[j] = x; s0s += x; s0q += x * x; }
#pragma unroll
    for (int j = 0; j < 12; ++j) { float x = vr[128 + tt * 12 + j]; a1[j] = x; s1q += x * x; }
#pragma unroll
    for (int j = 0; j < 10; ++j) { float x = vr[320 + tt * 10 + j]; a2[j] = x; s2q += x * x; }
    s0s = grp16_sum(s0s); s0q = grp16_sum(s0q);
    s1q = grp16_sum(s1q); s2q = grp16_sum(s2q);
    float mu0 = s0s * (1.f / NM0);
    float r0  = rsqrtf(s0q * (1.f / NM0) - mu0 * mu0 + 1e-5f);
    float r1  = rsqrtf(s1q * (1.f / (NM1 * 3)) + 1e-5f);
    float r2  = rsqrtf(s2q * (1.f / (NM2 * 5)) + 1e-5f);
#pragma unroll
    for (int j = 0; j < 8; ++j) {
      int c = tt * 8 + j;
      Asph[row * SPH + c] = (_Float16)((a0[j] - mu0) * r0 * o3w[c] + o3b[c]);
    }
#pragma unroll
    for (int j = 0; j < 12; ++j) {
      int f = tt * 12 + j; int i = f / 3, k = f % 3;
      Asph[row * SPH + 128 + k * 64 + i] = (_Float16)(a1[j] * r1 * o3w[NM0 + i]);
    }
#pragma unroll
    for (int j = 0; j < 10; ++j) {
      int f = tt * 10 + j; int i = f / 5, c5 = f % 5;
      Asph[row * SPH + 320 + c5 * 32 + i] = (_Float16)(a2[j] * r2 * o3w[NM0 + NM1 + i]);
    }
  }
  __syncthreads();

  // ---------------- U/V via WMMA (uniform per-segment loops) --------------
  // seg0: l=0, 16 tiles (U:8, V:8), K=128; each wave gets exactly 2 tiles.
#pragma unroll
  for (int t = wave; t < 16; t += 8) {
    int uv = t >> 3, r = t & 7;
    const _Float16* wT = uv ? Vw0T : Uw0T;
    int ocol = r * 16;
    v8f acc = {};
#pragma unroll
    for (int ks = 0; ks < 4; ++ks) {
      v16h a = *(const v16h*)(Asph + alo * SPH + ks * 32 + khalf);
      v16h b = *(const v16h*)(wT + (size_t)(ocol + alo) * 128 + ks * 32 + khalf);
      acc = WMMA_F16(a, b, acc);
    }
    float bias = (uv ? Vb0 : Ub0)[ocol + alo];
    float* obuf = uv ? Vbuf : Ubuf;
#pragma unroll
    for (int j = 0; j < 8; ++j)
      obuf[(j + mrow) * SPH + ocol + alo] = acc[j] + bias;
  }
  // seg1: l=1, 24 tiles (3 comps x 4 col-tiles x U/V), K=64; 3 tiles/wave.
#pragma unroll
  for (int t = wave; t < 24; t += 8) {
    int uv = t / 12, q = t % 12, k = q >> 2, n = q & 3;
    const _Float16* wT = uv ? Vw1T : Uw1T;
    int acol = 128 + k * 64, ocol = acol + n * 16;
    v8f acc = {};
#pragma unroll
    for (int ks = 0; ks < 2; ++ks) {
      v16h a = *(const v16h*)(Asph + alo * SPH + acol + ks * 32 + khalf);
      v16h b = *(const v16h*)(wT + (size_t)(n * 16 + alo) * 64 + ks * 32 + khalf);
      acc = WMMA_F16(a, b, acc);
    }
    float* obuf = uv ? Vbuf : Ubuf;
#pragma unroll
    for (int j = 0; j < 8; ++j)
      obuf[(j + mrow) * SPH + ocol + alo] = acc[j];
  }
  // seg2: l=2, 20 tiles (5 comps x 2 col-tiles x U/V), K=32.
  for (int t = wave; t < 20; t += 8) {
    int uv = t / 10, q = t % 10, c = q >> 1, n = q & 1;
    const _Float16* wT = uv ? Vw2T : Uw2T;
    int acol = 320 + c * 32, ocol = acol + n * 16;
    v16h a = *(const v16h*)(Asph + alo * SPH + acol + khalf);
    v16h b = *(const v16h*)(wT + (size_t)(n * 16 + alo) * 32 + khalf);
    v8f acc = {};
    acc = WMMA_F16(a, b, acc);
    float* obuf = uv ? Vbuf : Ubuf;
#pragma unroll
    for (int j = 0; j < 8; ++j)
      obuf[(j + mrow) * SPH + ocol + alo] = acc[j];
  }
  __syncthreads();

  // ---------------- invariants + dot features -----------------------------
  for (int idx = tid; idx < 16 * NIR; idx += 256) {
    int rr = idx / NIR, j = idx % NIR;
    float invv, dotv;
    if (j < 128) {
      float u = Ubuf[rr * SPH + j], v = Vbuf[rr * SPH + j];
      invv = fabsf(v); dotv = u * v;
    } else if (j < 192) {
      int o = j - 128; float ss = 0.f, dd = 0.f;
#pragma unroll
      for (int k = 0; k < 3; ++k) {
        float u = Ubuf[rr * SPH + 128 + k * 64 + o];
        float v = Vbuf[rr * SPH + 128 + k * 64 + o];
        ss += v * v; dd += u * v;
      }
      invv = sqrtf(ss); dotv = dd * 0.5773502691896258f;   // 1/sqrt(3)
    } else {
      int o = j - 192; float ss = 0.f, dd = 0.f;
#pragma unroll
      for (int c = 0; c < 5; ++c) {
        float u = Ubuf[rr * SPH + 320 + c * 32 + o];
        float v = Vbuf[rr * SPH + 320 + c * 32 + o];
        ss += v * v; dd += u * v;
      }
      invv = sqrtf(ss); dotv = dd * 0.4472135954999579f;   // 1/sqrt(5)
    }
    mlpA[rr * KMLP + 128 + j] = (_Float16)invv;
    dotA[rr * NIR + j]        = (_Float16)dotv;
  }
  __syncthreads();

  // ---------------- mlp1 (+silu) and inner = dot @ dot_w ------------------
  {
    const int n = wave * 16 + alo;
    v8f acc = {};
#pragma unroll
    for (int ks = 0; ks < 11; ++ks) {               // K = 352
      v16h a = *(const v16h*)(mlpA + alo * KMLP + ks * 32 + khalf);
      v16h b = *(const v16h*)(mw1T + (size_t)n * KMLP + ks * 32 + khalf);
      acc = WMMA_F16(a, b, acc);
    }
    float bias = mb1[n];
#pragma unroll
    for (int j = 0; j < 8; ++j) {
      float h = acc[j] + bias;
      h = h / (1.f + __expf(-h));                   // silu
      hA[(j + mrow) * NDIM + n] = (_Float16)h;
    }

    v8f acc2 = {};
#pragma unroll
    for (int ks = 0; ks < 7; ++ks) {                // K = 224
      v16h a = *(const v16h*)(dotA + alo * NIR + ks * 32 + khalf);
      v16h b = *(const v16h*)(dotwT + (size_t)n * NIR + ks * 32 + khalf);
      acc2 = WMMA_F16(a, b, acc2);
    }
#pragma unroll
    for (int j = 0; j < 8; ++j)
      innerB[(j + mrow) * NDIM + n] = acc2[j];
  }
  __syncthreads();

  // ---------------- mlp2: 16x128 @ 128x480 -> mlp_out ---------------------
  for (int t = wave; t < 30; t += 8) {
    int n0 = t * 16;
    v8f acc = {};
#pragma unroll
    for (int ks = 0; ks < 4; ++ks) {                // K = 128
      v16h a = *(const v16h*)(hA + alo * NDIM + ks * 32 + khalf);
      v16h b = *(const v16h*)(mw2T + (size_t)(n0 + alo) * NDIM + ks * 32 + khalf);
      acc = WMMA_F16(a, b, acc);
    }
    float bias = mb2[n0 + alo];
#pragma unroll
    for (int j = 0; j < 8; ++j)
      moutB[(j + mrow) * HID + n0 + alo] = acc[j] + bias;
  }
  __syncthreads();

  // ---------------- gates + residual writes -------------------------------
  for (int idx = tid; idx < 16 * NDIM; idx += 256) {
    int rr = idx >> 7, j = idx & 127;
    size_t rg = (size_t)blockIdx.x * 16 + rr;
    float a_sv = moutB[rr * HID + NIR + j];
    float a_ss = moutB[rr * HID + NIR + NDIM + j];
    out_s[rg * NDIM + j] = xs[rg * NDIM + j] + a_sv * innerB[rr * NDIM + j] + a_ss;
  }
  for (int idx = tid; idx < 16 * SPH; idx += 256) {
    int rr = idx / SPH, c = idx % SPH;
    size_t rg = (size_t)blockIdx.x * 16 + rr;
    float d;
    if (c < 128) {
      d = Ubuf[rr * SPH + c] * moutB[rr * HID + c];
    } else if (c < 320) {
      int f = c - 128, o = f / 3, k = f % 3;
      d = Ubuf[rr * SPH + 128 + k * 64 + o] * moutB[rr * HID + 128 + o];
    } else {
      int f = c - 320, o = f / 5, cc = f % 5;
      d = Ubuf[rr * SPH + 320 + cc * 32 + o] * moutB[rr * HID + 192 + o];
    }
    out_v[rg * SPH + c] = xv[rg * SPH + c] + d;
  }
}

extern "C" void kernel_launch(void* const* d_in, const int* in_sizes, int n_in,
                              void* d_out, int out_size, void* d_ws, size_t ws_size,
                              hipStream_t stream) {
  (void)in_sizes; (void)n_in; (void)out_size; (void)ws_size;
  const float* xs   = (const float*)d_in[0];
  const float* xv   = (const float*)d_in[1];
  const float* ln_g = (const float*)d_in[2];
  const float* ln_b = (const float*)d_in[3];
  const float* o3w  = (const float*)d_in[4];
  const float* o3b  = (const float*)d_in[5];
  const float* Uw0  = (const float*)d_in[6];
  const float* Uw1  = (const float*)d_in[7];
  const float* Uw2  = (const float*)d_in[8];
  const float* Ub0  = (const float*)d_in[9];
  const float* Vw0  = (const float*)d_in[10];
  const float* Vw1  = (const float*)d_in[11];
  const float* Vw2  = (const float*)d_in[12];
  const float* Vb0  = (const float*)d_in[13];
  const float* dotw = (const float*)d_in[14];
  const float* mw1  = (const float*)d_in[15];
  const float* mb1  = (const float*)d_in[16];
  const float* mw2  = (const float*)d_in[17];
  const float* mb2  = (const float*)d_in[18];

  // f16 transposed weight workspace layout (element offsets)
  _Float16* W     = (_Float16*)d_ws;
  _Float16* Uw0T  = W;              // 128*128 = 16384
  _Float16* Uw1T  = W + 16384;      // 64*64   = 4096
  _Float16* Uw2T  = W + 20480;      // 32*32   = 1024
  _Float16* Vw0T  = W + 21504;
  _Float16* Vw1T  = W + 37888;
  _Float16* Vw2T  = W + 41984;
  _Float16* dotwT = W + 43008;      // 224*128 = 28672
  _Float16* mw1T  = W + 71680;      // 352*128 = 45056
  _Float16* mw2T  = W + 116736;     // 128*480 = 61440  (end 178176 f16)

  auto cvt = [&](const float* src, _Float16* dst, int r, int c, float s) {
    int n = r * c;
    cvt_transpose_f16<<<(n + 255) / 256, 256, 0, stream>>>(src, dst, r, c, s);
  };
  const float s128 = 0.08838834764831845f;   // 1/sqrt(128)
  const float s64  = 0.125f;                 // 1/sqrt(64)
  const float s32  = 0.17677669529663687f;   // 1/sqrt(32)
  cvt(Uw0, Uw0T, 128, 128, s128);
  cvt(Uw1, Uw1T, 64, 64, s64);
  cvt(Uw2, Uw2T, 32, 32, s32);
  cvt(Vw0, Vw0T, 128, 128, s128);
  cvt(Vw1, Vw1T, 64, 64, s64);
  cvt(Vw2, Vw2T, 32, 32, s32);
  cvt(dotw, dotwT, 224, 128, 1.0f);
  cvt(mw1, mw1T, 352, 128, 1.0f);
  cvt(mw2, mw2T, 128, 480, 1.0f);

  float* out_s = (float*)d_out;
  float* out_v = out_s + (size_t)100000 * NDIM;

  xpainn_fused<<<6250, 256, SMEM_BYTES, stream>>>(
      xs, xv, ln_g, ln_b, o3w, o3b, Ub0, Vb0, mb1, mb2,
      Uw0T, Uw1T, Uw2T, Vw0T, Vw1T, Vw2T, dotwT, mw1T, mw2T,
      out_s, out_v);
}
